// Pose2Mesh_50440095924480
// MI455X (gfx1250) — matrix operand order repaired and checked by reading, not compiled, not run on hardware
//
#include <hip/hip_runtime.h>
#include <stdint.h>

// ---------------------------------------------------------------------------
// Pose2Mesh cheby-GCN forward for MI455X (gfx1250, wave32, WMMA).
//
// Bandwidth-bound network (~120 GFLOP vs multi-GB activation traffic at
// 23.3 TB/s). Design:
//   * All dense GEMMs via v_wmma_f32_16x16x32_bf16 (bf16 in, f32 acc).
//   * Cheby T0/T1/T2 packed directly to bf16 "xk" (M x Kpad) by the SpMM
//     kernels (halves GEMM A traffic vs fp32).
//   * BatchNorm(train-stats)+ReLU folded into consumers as per-channel
//     (scale, shift); GEMM epilogue accumulates sum/sumsq via LDS-reduced
//     global atomics; tiny kernel finalizes scale/shift.
//   * GEMM: 256 threads = 8 wave32 waves; macro tile 128M x 128N; wave tile
//     32M x 64N (8 WMMA per K-step); K-step 32; double-buffered LDS staging
//     via GLOBAL_LOAD_ASYNC_TO_LDS_B128 (ASYNCcnt-tracked, s_wait_asynccnt)
//     when the toolchain exposes the builtin, else synchronous b128 staging.
// ---------------------------------------------------------------------------

typedef __attribute__((ext_vector_type(16))) __bf16        v16bf;
typedef __attribute__((ext_vector_type(8)))  float         v8f;
typedef __attribute__((ext_vector_type(4)))  unsigned int  u32x4;
typedef __attribute__((ext_vector_type(4)))  int           i32x4;

typedef __attribute__((address_space(1))) i32x4 glb_i32x4;  // global v4i
typedef __attribute__((address_space(3))) i32x4 lds_i32x4;  // LDS v4i

#if defined(__has_builtin)
#if __has_builtin(__builtin_amdgcn_global_load_async_to_lds_b128) && \
    __has_builtin(__builtin_amdgcn_s_wait_asynccnt)
#define USE_ASYNC 1
#endif
#endif
#ifndef USE_ASYNC
#define USE_ASYNC 0
#endif

static __device__ __forceinline__ unsigned short f2bf(float f) {
  union { float f; unsigned u; } v; v.f = f;
  unsigned u = v.u;
  return (unsigned short)((u + 0x7FFFu + ((u >> 16) & 1u)) >> 16);  // RNE
}

static __device__ __forceinline__ float actv(float x, const float* sc,
                                             const float* sh, int f) {
  // sc==nullptr => tensor is already "final" (post-BN/ReLU or raw input)
  if (sc) x = fmaxf(x * sc[f] + sh[f], 0.0f);
  return x;
}

// ---------------------------------------------------------------------------
// Wt[Npad][Kpad] (bf16, zero-padded) = transpose(W[Krows][N])
// ---------------------------------------------------------------------------
__global__ void k_convert_wt(const float* __restrict__ W,
                             unsigned short* __restrict__ Wt,
                             int Krows, int N, int Kpad, int Npad) {
  int i = blockIdx.x * blockDim.x + threadIdx.x;
  if (i >= Npad * Kpad) return;
  int n = i / Kpad, k = i % Kpad;
  float v = (n < N && k < Krows) ? W[(size_t)k * N + n] : 0.0f;
  Wt[i] = f2bf(v);
}

// ---------------------------------------------------------------------------
// x1[b,v,f] = sum_e vals[v*8+e] * act(x[b, cols[v*8+e], f])      (T1 = L a)
// ---------------------------------------------------------------------------
__global__ void k_spmm1(const float* __restrict__ x, const int* __restrict__ cols,
                        const float* __restrict__ vals, const float* sc,
                        const float* sh, float* __restrict__ x1, int V, int F) {
  int f = blockIdx.x * blockDim.x + threadIdx.x;
  int v = blockIdx.y, b = blockIdx.z;
  if (f >= F) return;
  const float* xb = x + (size_t)b * V * F;
  float acc = 0.0f;
#pragma unroll
  for (int e = 0; e < 8; ++e) {
    int   c = cols[v * 8 + e];
    float w = vals[v * 8 + e];
    acc += w * actv(xb[(size_t)c * F + f], sc, sh, f);
  }
  x1[((size_t)b * V + v) * F + f] = acc;
}

// ---------------------------------------------------------------------------
// xk[(b*V+v), f*3 + {0,1,2}] = bf16{T0, T1, T2}, K-fastest; zero-pad to Kpad.
// T0 = act(x), T1 = x1, T2 = 2*L(x1) - T0.
// ---------------------------------------------------------------------------
__global__ void k_cheby_pack(const float* __restrict__ x,
                             const float* __restrict__ x1,
                             const int* __restrict__ cols,
                             const float* __restrict__ vals,
                             const float* sc, const float* sh,
                             unsigned short* __restrict__ xk,
                             int V, int F, int Kpad) {
  int f = blockIdx.x * blockDim.x + threadIdx.x;
  int v = blockIdx.y, b = blockIdx.z;
  size_t row = (size_t)b * V + v;
  if (f < F) {
    const float* xb  = x  + (size_t)b * V * F;
    const float* x1b = x1 + (size_t)b * V * F;
    float t0 = actv(xb[(size_t)v * F + f], sc, sh, f);
    float t1 = x1b[(size_t)v * F + f];
    float s  = 0.0f;
#pragma unroll
    for (int e = 0; e < 8; ++e) {
      int c = cols[v * 8 + e];
      s += vals[v * 8 + e] * x1b[(size_t)c * F + f];
    }
    float t2 = 2.0f * s - t0;
    unsigned short* p = xk + row * Kpad + (size_t)f * 3;
    p[0] = f2bf(t0); p[1] = f2bf(t1); p[2] = f2bf(t2);
  } else {
    int c = 3 * F + (f - F);            // zero K-padding (only Fin=3 needs it)
    if (c < Kpad) xk[row * Kpad + c] = 0;
  }
}

// ---------------------------------------------------------------------------
// FC input pack: flat (b,v,f) -> bf16 row b, col v*F+f (same flat order).
// ---------------------------------------------------------------------------
__global__ void k_pack_rows(const float* __restrict__ y, const float* sc,
                            const float* sh, unsigned short* __restrict__ xk,
                            size_t total, int F) {
  size_t i = blockIdx.x * (size_t)blockDim.x + threadIdx.x;
  if (i >= total) return;
  xk[i] = f2bf(actv(y[i], sc, sh, (int)(i % F)));
}

// ---------------------------------------------------------------------------
// Stage one 128x32 A tile + 128x32 B tile (bf16) into LDS.
// Async path: GLOBAL_LOAD_ASYNC_TO_LDS_B128, 4 ops/thread, ASYNCcnt-tracked.
// OOB A rows are clamped (results discarded by guarded stores/stats).
// ---------------------------------------------------------------------------
static __device__ __forceinline__ void stage_tiles(
    const unsigned short* __restrict__ A, const unsigned short* __restrict__ Bt,
    u32x4* lA, u32x4* lB, int tid, int m0, int n0, int k0, int M, int Kpad) {
#pragma unroll
  for (int r = 0; r < 2; ++r) {
    int idx = tid + (r << 8);            // 0..511
    int row = idx >> 2, kq = idx & 3;    // 128 rows x 4 uint4
    int gm = m0 + row;
    gm = gm < M ? gm : M - 1;            // clamp (fc and V=54 layers have ragged M)
    const void* ga = A  + (size_t)gm * Kpad         + (size_t)k0 + kq * 8;
    const void* gb = Bt + (size_t)(n0 + row) * Kpad + (size_t)k0 + kq * 8;
#if USE_ASYNC
    __builtin_amdgcn_global_load_async_to_lds_b128(
        (glb_i32x4*)ga, (lds_i32x4*)&lA[idx], 0, 0);
    __builtin_amdgcn_global_load_async_to_lds_b128(
        (glb_i32x4*)gb, (lds_i32x4*)&lB[idx], 0, 0);
#else
    __builtin_prefetch((const char*)ga + 64, 0, 0);
    lA[idx] = *(const u32x4*)ga;
    lB[idx] = *(const u32x4*)gb;
#endif
  }
}

// ---------------------------------------------------------------------------
// WMMA GEMM: C[M x N] = A[M x Kpad](bf16) @ Wt^T + bias, optional BN stats.
// ---------------------------------------------------------------------------
__global__ __launch_bounds__(256)
void k_wmma_gemm(const unsigned short* __restrict__ A,
                 const unsigned short* __restrict__ Bt,
                 const float* __restrict__ bias, float* __restrict__ C,
                 float* __restrict__ stats, int M, int N, int Kpad, int ldc,
                 int want_stats) {
  __shared__ u32x4 lA[2][512];    // 2 x (128 rows x 32 bf16)  = 16 KB
  __shared__ u32x4 lB[2][512];    // 2 x (128 rows x 32 bf16)  = 16 KB
  __shared__ float sSum[128], sSq[128];

  const int tid  = threadIdx.x;
  const int lane = tid & 31, wave = tid >> 5;
  const int waveM = wave & 3, waveN = wave >> 2;   // 4 x 2 wave grid
  const int h = lane >> 4, ml = lane & 15;
  const int m0 = blockIdx.x * 128, n0 = blockIdx.y * 128;

  if (tid < 128) { sSum[tid] = 0.0f; sSq[tid] = 0.0f; }

  v8f acc[2][4];
#pragma unroll
  for (int mf = 0; mf < 2; ++mf)
#pragma unroll
    for (int f = 0; f < 4; ++f)
#pragma unroll
      for (int i = 0; i < 8; ++i) acc[mf][f][i] = 0.0f;

  union Frag { u32x4 q[2]; v16bf v; };

  const int nsteps = Kpad >> 5;
  stage_tiles(A, Bt, lA[0], lB[0], tid, m0, n0, 0, M, Kpad);   // prologue

  for (int s = 0; s < nsteps; ++s) {
    const int buf = s & 1;
    if (s + 1 < nsteps) {
      stage_tiles(A, Bt, lA[buf ^ 1], lB[buf ^ 1], tid, m0, n0,
                  (s + 1) << 5, M, Kpad);
#if USE_ASYNC
      __builtin_amdgcn_s_wait_asynccnt(4);   // current buf's 4 copies done
#endif
    } else {
#if USE_ASYNC
      __builtin_amdgcn_s_wait_asynccnt(0);
#endif
    }
    __syncthreads();   // all waves' tile data visible

    const u32x4* pA = lA[buf];
    const u32x4* pB = lB[buf];
    // A frag (16x32): lane half h holds K in [8h,8h+8) and [16+8h,24+8h)
    Frag a[2];
#pragma unroll
    for (int mf = 0; mf < 2; ++mf) {
      int mr = waveM * 32 + mf * 16 + ml;
      a[mf].q[0] = pA[mr * 4 + h];
      a[mf].q[1] = pA[mr * 4 + 2 + h];
    }
#pragma unroll
    for (int f = 0; f < 4; ++f) {
      // B frag (32x16): lane n = ml, half h holds K in [8h,8h+8) and [16+8h,24+8h)
      Frag b;
      int nl = waveN * 64 + f * 16 + ml;
      b.q[0] = pB[nl * 4 + h];
      b.q[1] = pB[nl * 4 + 2 + h];
#pragma unroll
      for (int mf = 0; mf < 2; ++mf)
        acc[mf][f] = __builtin_amdgcn_wmma_f32_16x16x32_bf16(
            false, a[mf].v, false, b.v, (short)0, acc[mf][f], false, false);
    }
    __syncthreads();   // all waves done reading buf before it is restaged
  }

  // Epilogue: bias add, store, per-channel sum/sumsq (LDS reduce -> atomics).
#pragma unroll
  for (int f = 0; f < 4; ++f) {
    int nl = waveN * 64 + f * 16 + ml;
    int n  = n0 + nl;
    float bv = (n < N) ? bias[n] : 0.0f;
    float a1 = 0.0f, a2 = 0.0f;
#pragma unroll
    for (int mf = 0; mf < 2; ++mf)
#pragma unroll
      for (int r = 0; r < 8; ++r) {
        int row = m0 + waveM * 32 + mf * 16 + 8 * h + r;  // C/D: VGPR r<->M=r+8h
        float val = acc[mf][f][r] + bv;
        if (row < M && n < N) {
          C[(size_t)row * ldc + n] = val;
          a1 += val; a2 += val * val;
        }
      }
    if (want_stats) {
      atomicAdd(&sSum[nl], a1);
      atomicAdd(&sSq[nl],  a2);
    }
  }
  if (want_stats) {
    __syncthreads();
    if (tid < 128) {
      int gn = n0 + tid;
      if (gn < N) {
        atomicAdd(stats + gn,     sSum[tid]);
        atomicAdd(stats + N + gn, sSq[tid]);
      }
    }
  }
}

// ---------------------------------------------------------------------------
// scale = g*rsqrt(var+eps), shift = b - mean*scale  (biased var, train stats)
// ---------------------------------------------------------------------------
__global__ void k_bn_finalize(const float* __restrict__ stats,
                              const float* __restrict__ g,
                              const float* __restrict__ b, float* sc, float* sh,
                              int N, float invM) {
  int n = blockIdx.x * blockDim.x + threadIdx.x;
  if (n >= N) return;
  float mean = stats[n] * invM;
  float var  = stats[N + n] * invM - mean * mean;
  float s    = g[n] * rsqrtf(var + 1e-5f);
  sc[n] = s;
  sh[n] = b[n] - mean * s;
}

// ---------------------------------------------------------------------------
// out = interp_feat(inp, Cout) + relu(y*sc+sh); optional node repeat (x2).
// ---------------------------------------------------------------------------
__global__ void k_resid(const float* __restrict__ inp, const float* __restrict__ y,
                        const float* __restrict__ sc, const float* __restrict__ sh,
                        float* __restrict__ out, int V, int Cin, int Cout,
                        int do_repeat) {
  size_t i = blockIdx.x * (size_t)blockDim.x + threadIdx.x;
  size_t total = (size_t)32 * V * Cout;
  if (i >= total) return;
  int co = (int)(i % Cout);
  size_t bv = i / Cout;
  int v = (int)(bv % V);
  int b = (int)(bv / V);
  float r;
  if (Cin == Cout) {
    r = inp[i];
  } else {  // torch linear interp, align_corners=False
    float src = ((float)co + 0.5f) * ((float)Cin / (float)Cout) - 0.5f;
    src = fminf(fmaxf(src, 0.0f), (float)(Cin - 1));
    int lo = (int)src;
    int hi = min(lo + 1, Cin - 1);
    float w = src - (float)lo;
    const float* ip = inp + ((size_t)b * V + v) * Cin;
    r = ip[lo] * (1.0f - w) + ip[hi] * w;
  }
  float val = r + fmaxf(y[i] * sc[co] + sh[co], 0.0f);
  if (do_repeat) {
    float* op = out + ((size_t)b * (2 * V) + (size_t)(2 * v)) * Cout + co;
    op[0]    = val;
    op[Cout] = val;
  } else {
    out[i] = val;
  }
}

// ---------------------------------------------------------------------------
// Host orchestration
// ---------------------------------------------------------------------------
extern "C" void kernel_launch(void* const* d_in, const int* in_sizes, int n_in,
                              void* d_out, int out_size, void* d_ws,
                              size_t ws_size, hipStream_t stream) {
  (void)n_in; (void)out_size; (void)ws_size;

  // Input-order probe: insertion order => d_in[0] is x (32*24*3 = 2304);
  // JAX sorted-key pytree order => d_in[0] is graphs.cols[0] (24*8 = 192).
  const bool ins = (in_sizes[0] == 2304);
  auto CLW  = [&](int l) { return ins ? 1 + l   : 88 + l; };
  auto CLB  = [&](int l) { return ins ? 22 + l  : 67 + l; };
  auto BNG  = [&](int l) { return ins ? 43 + l  : 47 + l; };
  auto BNB  = [&](int l) { return ins ? 63 + l  : 27 + l; };
  auto COLS = [&](int g) { return ins ? 94 + g  : 0 + g;  };
  auto VALS = [&](int g) { return ins ? 103 + g : 18 + g; };
  const int FCW = ins ? 83 : 110, FCB = ins ? 84 : 109, XIN = ins ? 0 : 111;

  // Workspace carve-up (all 256B aligned).
  char* ws = (char*)d_ws;
  size_t off = 0;
  auto carve = [&](size_t bytes) -> char* {
    char* p = ws + off;
    off += (bytes + 255) & ~(size_t)255;
    return p;
  };
  const size_t ACT_B = (size_t)32 * 6912 * 128 * 4;      // 113.25 MB
  float* A0 = (float*)carve(ACT_B);
  float* A1 = (float*)carve(ACT_B);
  float* A2 = (float*)carve(ACT_B);
  float* X1 = (float*)carve(ACT_B);
  unsigned short* XK = (unsigned short*)carve((size_t)32 * 6912 * 384 * 2);
  unsigned short* WT = (unsigned short*)carve((size_t)3456 * 1536 * 2);
  float* STATS = (float*)carve((size_t)2 * 3456 * 4);
  float* SC = (float*)carve(4096);
  float* SH = (float*)carve(4096);

  auto run_cheby = [&](int l, int lvl, int V, int Fin, int Fout,
                       const float* xin, bool xAff, float* yout, int ldc,
                       bool bn) {
    const int Kr   = 3 * Fin;
    const int Kpad = ((Kr + 31) / 32) * 32;
    const int Npad = ((Fout + 127) / 128) * 128;
    const float* W    = (const float*)d_in[CLW(l)];
    const float* bias = (const float*)d_in[CLB(l)];
    const int*   cols = (const int*)d_in[COLS(lvl)];
    const float* vals = (const float*)d_in[VALS(lvl)];

    k_convert_wt<<<dim3((Npad * Kpad + 255) / 256), 256, 0, stream>>>(
        W, WT, Kr, Fout, Kpad, Npad);

    const float* sc = xAff ? SC : nullptr;
    const float* sh = xAff ? SH : nullptr;
    dim3 g1((unsigned)((Fin + 255) / 256), (unsigned)V, 32);
    k_spmm1<<<g1, 256, 0, stream>>>(xin, cols, vals, sc, sh, X1, V, Fin);
    k_cheby_pack<<<g1, 256, 0, stream>>>(xin, X1, cols, vals, sc, sh, XK, V,
                                         Fin, Kpad);

    const int M = 32 * V;
    if (bn) (void)hipMemsetAsync(STATS, 0, (size_t)2 * Fout * 4, stream);
    dim3 gg((unsigned)((M + 127) / 128), (unsigned)(Npad / 128));
    k_wmma_gemm<<<gg, 256, 0, stream>>>(XK, WT, bias, yout, STATS, M, Fout,
                                        Kpad, ldc, bn ? 1 : 0);
    if (bn)
      k_bn_finalize<<<dim3((Fout + 255) / 256), 256, 0, stream>>>(
          STATS, (const float*)d_in[BNG(l)], (const float*)d_in[BNB(l)], SC,
          SH, Fout, 1.0f / (float)M);
  };

  // ---- Block 0 (V=24): 3 -> 32 -> 64 -> 64 --------------------------------
  const float* x0 = (const float*)d_in[XIN];
  run_cheby(0, 0, 24, 3, 32, x0, false, A0, 32, true);
  run_cheby(1, 0, 24, 32, 64, A0, true, A1, 64, true);
  run_cheby(2, 0, 24, 64, 64, A1, true, A0, 64, true);

  // ---- FC: (32, 24*64) @ (1536, 3456) -> (32, 54, 64), no BN --------------
  {
    size_t tot = (size_t)32 * 24 * 64;
    k_pack_rows<<<dim3((unsigned)((tot + 255) / 256)), 256, 0, stream>>>(
        A0, SC, SH, XK, tot, 64);
    k_convert_wt<<<dim3((3456 * 1536 + 255) / 256), 256, 0, stream>>>(
        (const float*)d_in[FCW], WT, 1536, 3456, 1536, 3456);
    k_wmma_gemm<<<dim3(1, 27), 256, 0, stream>>>(
        XK, WT, (const float*)d_in[FCB], A1, STATS, 32, 3456, 1536, 3456, 0);
  }

  // ---- Blocks 1..8: 2 cheby layers + residual (+repeat for 1..7) ----------
  struct Blk { int lvl, V, Cin, Fm, Fo, l0, rep; };
  const Blk blks[8] = {
      {1,   54,  64, 128, 256,  3, 1},
      {2,  108, 256, 256, 256,  5, 1},
      {3,  216, 256, 256, 256,  7, 1},
      {4,  432, 256, 256, 256,  9, 1},
      {5,  864, 256, 256, 256, 11, 1},
      {6, 1728, 256, 128, 128, 13, 1},
      {7, 3456, 128, 128, 128, 15, 1},
      {8, 6912, 128, 128, 128, 17, 0},
  };
  float* xin = A1;  // block input (always post-activation "final" values)
  float* f0  = A0;
  float* f1  = A2;
  for (int i = 0; i < 8; ++i) {
    const Blk& K = blks[i];
    run_cheby(K.l0,     K.lvl, K.V, K.Cin, K.Fm, xin, false, f0, K.Fm, true);
    run_cheby(K.l0 + 1, K.lvl, K.V, K.Fm,  K.Fo, f0,  true,  f1, K.Fo, true);
    size_t tot = (size_t)32 * K.V * K.Fo;
    k_resid<<<dim3((unsigned)((tot + 255) / 256)), 256, 0, stream>>>(
        xin, f1, SC, SH, f0, K.V, K.Cin, K.Fo, K.rep);
    float* t = xin; xin = f0; f0 = t;  // f1 reused as scratch next block
  }

  // ---- Block 9 (V=6912): 128 -> 64 -> 3, last layer no BN, write d_out ----
  run_cheby(19, 8, 6912, 128, 64, xin, false, f0, 64, true);
  run_cheby(20, 8, 6912, 64, 3, f0, true, (float*)d_out, 3, false);
}
